// Struct2SeqGCN_30167850287447
// MI455X (gfx1250) — matrix operand order repaired and checked by reading, not compiled, not run on hardware
//
#include <hip/hip_runtime.h>
#include <hip/hip_bf16.h>

typedef __attribute__((ext_vector_type(16))) _Float16 v16h;
typedef __attribute__((ext_vector_type(8)))  float    v8f;

#define N_NODES   25000
#define N_EDGES   400000
#define F_IN      6
#define HDIM      128
#define NCLS      21
#define ZDIM      272              // 2*H + 16 gaussians
#define ZPAD      288              // padded K = 9 * 32
#define KTILES    9
#define NTILES    8
#define TILE_E    32               // edges per block (2 WMMA M-subtiles)
#define FRAG_HALVES (KTILES*NTILES*32*16)   // 36864 f16 per weight matrix

// ---------------------------------------------------------------- utilities
__global__ __launch_bounds__(256) void fill_f32(float* __restrict__ p, int n, float v) {
  int i = blockIdx.x * 256 + threadIdx.x;
  if (i < n) p[i] = v;
}

// ------------------------------------------------------------- node embed
// h = x @ node_w + node_b  ([25000,6]@[6,128]); also writes the f16 mirror
// that feeds the async-LDS staging in the edge kernel.
__global__ __launch_bounds__(256) void node_embed(const float* __restrict__ x,
    const float* __restrict__ w, const float* __restrict__ b,
    float* __restrict__ h, void* __restrict__ h16v) {
  _Float16* h16 = (_Float16*)h16v;
  int i = blockIdx.x * 256 + threadIdx.x;
  if (i >= N_NODES * HDIM) return;
  int n = i / HDIM, c = i % HDIM;
  float acc = b[c];
#pragma unroll
  for (int f = 0; f < F_IN; ++f) acc += x[n * F_IN + f] * w[f * HDIM + c];
  h[i] = acc;
  h16[i] = (_Float16)acc;
}

// ---------------------------------------------------- weight pre-swizzle
// [272,128] f32 -> f16 WMMA B-operand fragments, K padded to 288.
// B k order used here (contiguous; matches the A loads): lanes 0-15 K=0..15 (k=elem),
// lanes 16-31 K=16..31 (k=16+elem); n = lane&15.  Each lane's 16 halves are
// one contiguous 32-byte vector.
__global__ __launch_bounds__(256) void prep_frags(const float* __restrict__ w0,
    const float* __restrict__ w1, const float* __restrict__ w2,
    const float* __restrict__ w3, void* fragv) {
  _Float16* frag = (_Float16*)fragv;
  int i = blockIdx.x * 256 + threadIdx.x;
  if (i >= 4 * FRAG_HALVES) return;
  int mat  = i / FRAG_HALVES;
  int rem  = i - mat * FRAG_HALVES;
  int t    = rem >> 9;
  int le   = rem & 511;
  int lane = le >> 4;
  int e    = le & 15;
  int kt = t >> 3, nt = t & 7;
  int half = lane >> 4;
  int k = kt * 32 + half * 16 + e;
  int n = nt * 16 + (lane & 15);
  const float* w = (mat == 0) ? w0 : (mat == 1) ? w1 : (mat == 2) ? w2 : w3;
  float v = (k < ZDIM) ? w[k * HDIM + n] : 0.0f;
  frag[i] = (_Float16)v;
}

// ----------------------------------------------------- edge message kernel
// Block = 32 edges (2 WMMA M-subtiles), 8 waves; wave w owns N-columns
// [16w,16w+16).  Node-feature rows are DMA'd f16 from the h16 mirror into
// LDS with GLOBAL_LOAD_ASYNC_TO_LDS_B128 (ASYNCcnt); gaussians computed by
// VALU.  Then 9 k-steps x {2 subtiles x 2 matrices} of wmma_f32_16x16x32_f16
// sharing each B-fragment pair, sigmoid*softplus gate, atomic scatter-add.
__global__ __launch_bounds__(256) void edge_messages(
    const void* __restrict__ h16v,
    const int* __restrict__ srcI, const int* __restrict__ dstI,
    const float* __restrict__ ea,
    const void* __restrict__ fragFv, const void* __restrict__ fragSv,
    const float* __restrict__ fb, const float* __restrict__ sb,
    float* __restrict__ agg) {
  const _Float16* __restrict__ h16 = (const _Float16*)h16v;
  const v16h* __restrict__ fragF = (const v16h*)fragFv;
  const v16h* __restrict__ fragS = (const v16h*)fragSv;

  __shared__ __align__(16) _Float16 zs[TILE_E][ZPAD];  // 18432 B
  __shared__ int dstL[TILE_E];
  __shared__ int srcL[TILE_E];

  const int tid = threadIdx.x;
  const int e0  = blockIdx.x * TILE_E;

  if (tid < TILE_E)           dstL[tid]          = dstI[e0 + tid];
  else if (tid < 2 * TILE_E)  srcL[tid - TILE_E] = srcI[e0 + tid - TILE_E];
  __syncthreads();

  // Gaussian smearing + zero pad (VALU; 32 edges x 16 each).
  for (int q = tid; q < TILE_E * 16; q += 256) {
    int m = q >> 4, j = q & 15;
    float mu = (8.0f / 15.0f) * (float)j;
    float d  = ea[e0 + m] - mu;
    zs[m][2 * HDIM + j] = (_Float16)__expf(-1.7578125f * d * d);
    zs[m][ZDIM + j]     = (_Float16)0.0f;   // pad 272..287
  }

  // Async DMA: 32 edges x {h16[dst] row, h16[src] row} x 256B, 16B per xfer.
  // Generic LDS pointers carry the allocation-relative offset in addr[31:0].
  const unsigned int zbase = (unsigned int)(size_t)(&zs[0][0]);
  for (int q = tid; q < TILE_E * 2 * 16; q += 256) {
    int r  = q >> 4;                 // 0..63 : edge row
    int ch = q & 15;                 // 16B chunk within the 256B row
    int m  = r >> 1, wh = r & 1;     // wh: 0 = dst half, 1 = src half
    int node = wh ? srcL[m] : dstL[m];
    const _Float16* gp = h16 + (size_t)node * HDIM + ch * 8;
    unsigned int lds = zbase + (unsigned int)(m * (ZPAD * 2) + wh * 256 + ch * 16);
    asm volatile("global_load_async_to_lds_b128 %0, %1, off"
                 :: "v"(lds), "v"(gp) : "memory");
  }
  asm volatile("s_wait_asynccnt 0x0" ::: "memory");
  __syncthreads();

  const int lane = tid & 31;
  const int wv   = tid >> 5;         // N-tile owned by this wave
  const int half = lane >> 4;
  const int nl   = lane & 15;
  const int n    = wv * 16 + nl;

  v8f aF0, aS0, aF1, aS1;
  const float bf = fb[n], bs = sb[n];
#pragma unroll
  for (int r = 0; r < 8; ++r) { aF0[r] = bf; aS0[r] = bs; aF1[r] = bf; aS1[r] = bs; }

#pragma unroll
  for (int kt = 0; kt < KTILES; ++kt) {
    // A k order used here (matches B frags): lane row M = lane&15; lane's k values are
    // {kt*32 + 16*half + 0..15}: two consecutive aligned 16B LDS chunks.
    union { v16h v; uint4 u[2]; } a0, a1;
    const uint4* p0 = reinterpret_cast<const uint4*>(&zs[nl][kt * 32 + half * 16]);
    const uint4* p1 = reinterpret_cast<const uint4*>(&zs[16 + nl][kt * 32 + half * 16]);
    a0.u[0] = p0[0]; a0.u[1] = p0[1];
    a1.u[0] = p1[0]; a1.u[1] = p1[1];
    const v16h bF = fragF[(kt * NTILES + wv) * 32 + lane];
    const v16h bS = fragS[(kt * NTILES + wv) * 32 + lane];
    aF0 = __builtin_amdgcn_wmma_f32_16x16x32_f16(false, a0.v, false, bF, (short)0, aF0, false, false);
    aS0 = __builtin_amdgcn_wmma_f32_16x16x32_f16(false, a0.v, false, bS, (short)0, aS0, false, false);
    aF1 = __builtin_amdgcn_wmma_f32_16x16x32_f16(false, a1.v, false, bF, (short)0, aF1, false, false);
    aS1 = __builtin_amdgcn_wmma_f32_16x16x32_f16(false, a1.v, false, bS, (short)0, aS1, false, false);
  }

  // C/D layout: VGPR r, lanes<16 -> M=r, lanes>=16 -> M=r+8; N = lane&15.
#pragma unroll
  for (int r = 0; r < 8; ++r) {
    int m0 = r + half * 8;
    int m1 = 16 + m0;
    {
      float f = aF0[r], s = aS0[r];
      float sig = 1.0f / (1.0f + __expf(-f));
      float sp  = (s > 20.0f) ? s : __logf(1.0f + __expf(s));
      atomicAdd(&agg[dstL[m0] * HDIM + n], sig * sp);
    }
    {
      float f = aF1[r], s = aS1[r];
      float sig = 1.0f / (1.0f + __expf(-f));
      float sp  = (s > 20.0f) ? s : __logf(1.0f + __expf(s));
      atomicAdd(&agg[dstL[m1] * HDIM + n], sig * sp);
    }
  }
}

// ---------------------------------------------------------- batch-norm
__global__ __launch_bounds__(128) void bn_stats(const float* __restrict__ agg,
                                                float* __restrict__ stats) {
  int c  = threadIdx.x;
  int r0 = blockIdx.x * 125;
  int r1 = r0 + 125; if (r1 > N_NODES) r1 = N_NODES;
  float s = 0.0f, s2 = 0.0f;
  for (int r = r0; r < r1; ++r) {
    float v = agg[r * HDIM + c];
    s += v; s2 += v * v;
  }
  atomicAdd(&stats[c], s);
  atomicAdd(&stats[HDIM + c], s2);
}

__global__ __launch_bounds__(256) void bn_apply_residual_relu(
    const float* __restrict__ agg, float* __restrict__ h,
    void* __restrict__ h16v,
    const float* __restrict__ stats,
    const float* __restrict__ g, const float* __restrict__ b) {
  _Float16* h16 = (_Float16*)h16v;
  int i = blockIdx.x * 256 + threadIdx.x;
  if (i >= N_NODES * HDIM) return;
  int c = i % HDIM;
  float mean = stats[c] * (1.0f / N_NODES);
  float var  = stats[HDIM + c] * (1.0f / N_NODES) - mean * mean;
  float y = (agg[i] - mean) * rsqrtf(var + 1e-5f) * g[c] + b[c] + h[i];
  y = fmaxf(y, 0.0f);
  h[i] = y;
  h16[i] = (_Float16)y;
}

// ---------------------------------------------------------- final FC
__global__ __launch_bounds__(256) void fc_out(const float* __restrict__ h,
    const float* __restrict__ w, const float* __restrict__ b,
    float* __restrict__ out) {
  int i = blockIdx.x * 256 + threadIdx.x;
  if (i >= N_NODES * NCLS) return;
  int n = i / NCLS, j = i % NCLS;
  float acc = b[j];
  const float* hr = h + n * HDIM;
#pragma unroll 4
  for (int c = 0; c < HDIM; ++c) acc += hr[c] * w[c * NCLS + j];
  out[i] = acc;
}

// ---------------------------------------------------------------- driver
extern "C" void kernel_launch(void* const* d_in, const int* in_sizes, int n_in,
                              void* d_out, int out_size, void* d_ws, size_t ws_size,
                              hipStream_t stream) {
  const float* x      = (const float*)d_in[0];
  const float* ea     = (const float*)d_in[1];
  const float* node_w = (const float*)d_in[2];
  const float* node_b = (const float*)d_in[3];
  const float* f1_w   = (const float*)d_in[4];
  const float* f1_b   = (const float*)d_in[5];
  const float* s1_w   = (const float*)d_in[6];
  const float* s1_b   = (const float*)d_in[7];
  const float* bn1_g  = (const float*)d_in[8];
  const float* bn1_b  = (const float*)d_in[9];
  const float* f2_w   = (const float*)d_in[10];
  const float* f2_b   = (const float*)d_in[11];
  const float* s2_w   = (const float*)d_in[12];
  const float* s2_b   = (const float*)d_in[13];
  const float* bn2_g  = (const float*)d_in[14];
  const float* bn2_b  = (const float*)d_in[15];
  const float* fc_w   = (const float*)d_in[16];
  const float* fc_b   = (const float*)d_in[17];
  const int*   eidx   = (const int*)d_in[18];
  const int* srcI = eidx;            // edge_index[0]
  const int* dstI = eidx + N_EDGES;  // edge_index[1]
  float* out = (float*)d_out;

  // Workspace layout (~32.3 MB)
  char*  ws    = (char*)d_ws;
  float* hA    = (float*)ws;                                    // f32 features
  float* hB    = (float*)(ws + (size_t)N_NODES * HDIM * 4);     // agg buffer
  void*  h16   = (void*)(ws + (size_t)2 * N_NODES * HDIM * 4);  // f16 mirror
  char*  frag  = ws + (size_t)2 * N_NODES * HDIM * 4 + (size_t)N_NODES * HDIM * 2;
  float* stats = (float*)(frag + (size_t)4 * FRAG_HALVES * 2);
  const void* fragF1 = frag;
  const void* fragS1 = frag + (size_t)1 * FRAG_HALVES * 2;
  const void* fragF2 = frag + (size_t)2 * FRAG_HALVES * 2;
  const void* fragS2 = frag + (size_t)3 * FRAG_HALVES * 2;

  const int NH = N_NODES * HDIM;

  prep_frags<<<(4 * FRAG_HALVES + 255) / 256, 256, 0, stream>>>(
      f1_w, s1_w, f2_w, s2_w, frag);
  node_embed<<<(NH + 255) / 256, 256, 0, stream>>>(x, node_w, node_b, hA, h16);

  // ---- layer 1 ----
  fill_f32<<<(NH + 255) / 256, 256, 0, stream>>>(hB, NH, 0.0f);
  fill_f32<<<1, 256, 0, stream>>>(stats, 2 * HDIM, 0.0f);
  edge_messages<<<N_EDGES / TILE_E, 256, 0, stream>>>(
      h16, srcI, dstI, ea, fragF1, fragS1, f1_b, s1_b, hB);
  bn_stats<<<(N_NODES + 124) / 125, 128, 0, stream>>>(hB, stats);
  bn_apply_residual_relu<<<(NH + 255) / 256, 256, 0, stream>>>(
      hB, hA, h16, stats, bn1_g, bn1_b);

  // ---- layer 2 ----
  fill_f32<<<(NH + 255) / 256, 256, 0, stream>>>(hB, NH, 0.0f);
  fill_f32<<<1, 256, 0, stream>>>(stats, 2 * HDIM, 0.0f);
  edge_messages<<<N_EDGES / TILE_E, 256, 0, stream>>>(
      h16, srcI, dstI, ea, fragF2, fragS2, f2_b, s2_b, hB);
  bn_stats<<<(N_NODES + 124) / 125, 128, 0, stream>>>(hB, stats);
  bn_apply_residual_relu<<<(NH + 255) / 256, 256, 0, stream>>>(
      hB, hA, h16, stats, bn2_g, bn2_b);

  fc_out<<<(N_NODES * NCLS + 255) / 256, 256, 0, stream>>>(hA, fc_w, fc_b, out);
}
